// SelfLoss_86096914415965
// MI455X (gfx1250) — matrix operand order repaired and checked by reading, not compiled, not run on hardware
//
#include <hip/hip_runtime.h>

// Fused SelfLoss for MI455X (gfx1250, wave32, WMMA).
// GEMMs run as bf16 hi/lo split (3x v_wmma_f32_16x16x32_bf16 per K=32 block,
// ~fp32 accuracy at ~2.7x the native f32-WMMA rate). All row reductions are
// fused into the GEMM epilogue so the 164MB cos matrix is never materialized.
// Workspace requirement: ~112 MiB (hi/lo planes + partial stats).

#define ALPHA_F 10.0f
#define BETA_F  2.0f

constexpr int Nn = 4096;
constexpr int Cc = 10000;
constexpr int Dd = 2048;

constexpr int ROWS_BLK   = 64;              // rows per WG (4 waves x 16)
constexpr int CHUNK_COLS = 400;             // cols per WG in gemm1
constexpr int NCHUNK     = Cc / CHUNK_COLS; // 25 column chunks
constexpr int CT_GRP     = 5;               // 16-col tiles resident in regs
constexpr int SIM_TILES  = Nn / ROWS_BLK;   // 64

typedef __attribute__((ext_vector_type(16))) __bf16 v16bf;
typedef __attribute__((ext_vector_type(8)))  float  v8f;

union FragB16 { uint4 q[2]; v16bf v; };

__device__ __forceinline__ v8f zero8() {
  v8f z = {0.f,0.f,0.f,0.f,0.f,0.f,0.f,0.f};
  return z;
}

__device__ __forceinline__ unsigned short f32_to_bf16(float f) {
  unsigned u = __float_as_uint(f);
  u += 0x7FFFu + ((u >> 16) & 1u);     // round to nearest even
  return (unsigned short)(u >> 16);
}
__device__ __forceinline__ float bf16_to_f32(unsigned short h) {
  return __uint_as_float(((unsigned)h) << 16);
}
// order-preserving float -> uint key (monotone under unsigned compare)
__device__ __forceinline__ unsigned fkey(float f) {
  unsigned u = __float_as_uint(f);
  return (u & 0x80000000u) ? ~u : (u | 0x80000000u);
}
__device__ __forceinline__ float unfkey(unsigned k) {
  unsigned u = (k & 0x80000000u) ? (k ^ 0x80000000u) : ~k;
  return __uint_as_float(u);
}
__device__ __forceinline__ v8f wmma_bf16(v16bf a, v16bf b, v8f c) {
  return __builtin_amdgcn_wmma_f32_16x16x32_bf16(
      /*neg_a=*/false, a, /*neg_b=*/false, b,
      /*c_mod=*/(short)0, c, /*reuse_a=*/false, /*reuse_b=*/false);
}

// ---------------------------------------------------------------- split f32 -> bf16 hi/lo
__global__ __launch_bounds__(256) void split_bf16_kernel(
    const float* __restrict__ src, unsigned short* __restrict__ hi,
    unsigned short* __restrict__ lo, int n) {
  int idx = blockIdx.x * blockDim.x + threadIdx.x;
  int stride = gridDim.x * blockDim.x;
  for (int i = idx; i < n; i += stride) {
    float x = src[i];
    unsigned short h = f32_to_bf16(x);
    float r = x - bf16_to_f32(h);
    hi[i] = h;
    lo[i] = f32_to_bf16(r);
  }
}

// ---------------------------------------------------------------- phi = <inputs[i], kernel[t_i]> - beta (exact f32)
__global__ __launch_bounds__(32) void phi_kernel(
    const float* __restrict__ A, const float* __restrict__ K,
    const int* __restrict__ targets, float* __restrict__ phiArr) {
  int row = blockIdx.x;
  int lane = threadIdx.x;
  int t = targets[row];
  const float4* a4 = reinterpret_cast<const float4*>(A + (size_t)row * Dd);
  const float4* k4 = reinterpret_cast<const float4*>(K + (size_t)t * Dd);
  float s = 0.f;
  for (int i = lane; i < Dd / 4; i += 32) {
    float4 av = a4[i], kv = k4[i];
    s += av.x * kv.x + av.y * kv.y + av.z * kv.z + av.w * kv.w;
  }
  #pragma unroll
  for (int m = 16; m >= 1; m >>= 1) s += __shfl_xor(s, m, 32);
  if (lane == 0) phiArr[row] = s - BETA_F;
}

// ---------------------------------------------------------------- GEMM1 + fused row stats
// grid = (Nn/64, 25 chunks), block = 128 (4 waves; wave w owns 16 rows)
__global__ __launch_bounds__(128) void gemm1_kernel(
    const unsigned short* __restrict__ Ahi, const unsigned short* __restrict__ Alo,
    const unsigned short* __restrict__ Bhi, const unsigned short* __restrict__ Blo,
    const int* __restrict__ targets, float* __restrict__ rowstats) {
  const int wave = threadIdx.x >> 5, lane = threadIdx.x & 31;
  const int khalf = lane >> 4, lcol = lane & 15;
  const int rowBase = blockIdx.x * ROWS_BLK + wave * 16;
  const int colBase = blockIdx.y * CHUNK_COLS;

  const int arow = rowBase + lcol;                      // A-matrix: M = lane&15
  const unsigned short* aHiRow = Ahi + (size_t)arow * Dd;
  const unsigned short* aLoRow = Alo + (size_t)arow * Dd;

  int tg[8];
  #pragma unroll
  for (int j = 0; j < 8; ++j) tg[j] = targets[rowBase + khalf * 8 + j];

  float se[8], cs[8]; unsigned bk[8]; int bc[8];
  #pragma unroll
  for (int j = 0; j < 8; ++j) { se[j] = 0.f; cs[j] = 0.f; bk[j] = 0u; bc[j] = 0; }

  for (int g = 0; g < (CHUNK_COLS / 16) / CT_GRP; ++g) {   // 5 groups of 5 tiles
    v8f acc[CT_GRP];
    #pragma unroll
    for (int c = 0; c < CT_GRP; ++c) acc[c] = zero8();

    for (int kb = 0; kb < Dd / 32; ++kb) {
      const int k0 = kb * 32 + khalf * 8;                  // A: two 8-elt chunks
      FragB16 ahi, alo;
      ahi.q[0] = *(const uint4*)(aHiRow + k0);
      ahi.q[1] = *(const uint4*)(aHiRow + k0 + 16);
      alo.q[0] = *(const uint4*)(aLoRow + k0);
      alo.q[1] = *(const uint4*)(aLoRow + k0 + 16);
      const int kB = kb * 32 + khalf * 8;                  // B: two 8-elt chunks
      #pragma unroll
      for (int c = 0; c < CT_GRP; ++c) {
        const int n = colBase + (g * CT_GRP + c) * 16 + lcol;
        const unsigned short* bH = Bhi + (size_t)n * Dd + kB;
        const unsigned short* bL = Blo + (size_t)n * Dd + kB;
        FragB16 bhi, blo;
        bhi.q[0] = ((const uint4*)bH)[0]; bhi.q[1] = ((const uint4*)bH)[2];
        blo.q[0] = ((const uint4*)bL)[0]; blo.q[1] = ((const uint4*)bL)[2];
        acc[c] = wmma_bf16(alo.v, bhi.v, acc[c]);
        acc[c] = wmma_bf16(ahi.v, blo.v, acc[c]);
        acc[c] = wmma_bf16(ahi.v, bhi.v, acc[c]);
      }
    }
    // fused epilogue: clip, exp-sum (excluding target col), cos-sum, max/argmax
    #pragma unroll
    for (int c = 0; c < CT_GRP; ++c) {
      const int col = colBase + (g * CT_GRP + c) * 16 + lcol;
      #pragma unroll
      for (int j = 0; j < 8; ++j) {
        float v = acc[c][j];
        float cv = fminf(1.f, fmaxf(-1.f, v));
        bool iseq = (col == tg[j]);
        float outv = iseq ? (v - BETA_F) : cv;
        cs[j] += cv;
        se[j] += iseq ? 0.f : __expf(ALPHA_F * cv);
        unsigned ok = fkey(outv);
        if (ok > bk[j] || (ok == bk[j] && col < bc[j])) { bk[j] = ok; bc[j] = col; }
      }
    }
  }
  // reduce over the 16 lanes sharing each row (xor within half-wave)
  #pragma unroll
  for (int m = 1; m < 16; m <<= 1) {
    #pragma unroll
    for (int j = 0; j < 8; ++j) {
      se[j] += __shfl_xor(se[j], m, 32);
      cs[j] += __shfl_xor(cs[j], m, 32);
      unsigned ok = __shfl_xor(bk[j], m, 32);
      int      oc = __shfl_xor(bc[j], m, 32);
      if (ok > bk[j] || (ok == bk[j] && oc < bc[j])) { bk[j] = ok; bc[j] = oc; }
    }
  }
  if (lcol == 0) {
    #pragma unroll
    for (int j = 0; j < 8; ++j) {
      int row = rowBase + khalf * 8 + j;
      float* p = rowstats + ((size_t)row * NCHUNK + blockIdx.y) * 4;
      p[0] = se[j]; p[1] = cs[j];
      p[2] = __uint_as_float(bk[j]); p[3] = __int_as_float(bc[j]);
    }
  }
}

// ---------------------------------------------------------------- sim = inputs @ inputs.T with fused masked sums
// grid = (64, 64) tiles of 64x64, block = 128
__global__ __launch_bounds__(128) void sim_kernel(
    const unsigned short* __restrict__ Ahi, const unsigned short* __restrict__ Alo,
    const int* __restrict__ targets, float* __restrict__ part) {
  const int wave = threadIdx.x >> 5, lane = threadIdx.x & 31;
  const int khalf = lane >> 4, lcol = lane & 15;
  const int rowBase = blockIdx.x * ROWS_BLK + wave * 16;
  const int colBase = blockIdx.y * ROWS_BLK;

  const int arow = rowBase + lcol;
  const unsigned short* aHiRow = Ahi + (size_t)arow * Dd;
  const unsigned short* aLoRow = Alo + (size_t)arow * Dd;

  int tg[8];
  #pragma unroll
  for (int j = 0; j < 8; ++j) tg[j] = targets[rowBase + khalf * 8 + j];
  int tc[4];
  #pragma unroll
  for (int c = 0; c < 4; ++c) tc[c] = targets[colBase + c * 16 + lcol];

  v8f acc[4];
  #pragma unroll
  for (int c = 0; c < 4; ++c) acc[c] = zero8();

  for (int kb = 0; kb < Dd / 32; ++kb) {
    const int k0 = kb * 32 + khalf * 8;
    FragB16 ahi, alo;
    ahi.q[0] = *(const uint4*)(aHiRow + k0);
    ahi.q[1] = *(const uint4*)(aHiRow + k0 + 16);
    alo.q[0] = *(const uint4*)(aLoRow + k0);
    alo.q[1] = *(const uint4*)(aLoRow + k0 + 16);
    const int kB = kb * 32 + khalf * 8;
    #pragma unroll
    for (int c = 0; c < 4; ++c) {
      const int n = colBase + c * 16 + lcol;
      const unsigned short* bH = Ahi + (size_t)n * Dd + kB;
      const unsigned short* bL = Alo + (size_t)n * Dd + kB;
      FragB16 bhi, blo;
      bhi.q[0] = ((const uint4*)bH)[0]; bhi.q[1] = ((const uint4*)bH)[2];
      blo.q[0] = ((const uint4*)bL)[0]; blo.q[1] = ((const uint4*)bL)[2];
      acc[c] = wmma_bf16(alo.v, bhi.v, acc[c]);
      acc[c] = wmma_bf16(ahi.v, blo.v, acc[c]);
      acc[c] = wmma_bf16(ahi.v, bhi.v, acc[c]);
    }
  }
  float tot = 0.f, pos = 0.f, cnt = 0.f;
  #pragma unroll
  for (int c = 0; c < 4; ++c) {
    #pragma unroll
    for (int j = 0; j < 8; ++j) {
      float s = acc[c][j];
      bool eq = (tg[j] == tc[c]);
      tot += s;
      pos += eq ? s : 0.f;
      cnt += eq ? 1.f : 0.f;
    }
  }
  #pragma unroll
  for (int m = 16; m >= 1; m >>= 1) {
    tot += __shfl_xor(tot, m, 32);
    pos += __shfl_xor(pos, m, 32);
    cnt += __shfl_xor(cnt, m, 32);
  }
  __shared__ float sp[4][3];
  if (lane == 0) { sp[wave][0] = pos; sp[wave][1] = tot; sp[wave][2] = cnt; }
  __syncthreads();
  if (threadIdx.x == 0) {
    float P = sp[0][0] + sp[1][0] + sp[2][0] + sp[3][0];
    float T = sp[0][1] + sp[1][1] + sp[2][1] + sp[3][1];
    float Cn = sp[0][2] + sp[1][2] + sp[2][2] + sp[3][2];
    size_t bid = (size_t)blockIdx.x * gridDim.y + blockIdx.y;
    part[bid * 3 + 0] = P; part[bid * 3 + 1] = T; part[bid * 3 + 2] = Cn;
  }
}

// ---------------------------------------------------------------- final deterministic reduction -> 6 scalars
__global__ __launch_bounds__(256) void finalize_kernel(
    const float* __restrict__ rowstats, const float* __restrict__ phiArr,
    const int* __restrict__ targets, const float* __restrict__ simpart,
    float* __restrict__ out) {
  int tid = threadIdx.x;
  float losss = 0.f, maskc = 0.f, precc = 0.f, poss = 0.f, costot = 0.f;
  for (int row = tid; row < Nn; row += 256) {
    float se = 0.f, cst = 0.f; unsigned bk = 0u; int bc = 0;
    for (int ch = 0; ch < NCHUNK; ++ch) {
      const float* p = rowstats + ((size_t)row * NCHUNK + ch) * 4;
      se += p[0]; cst += p[1];
      unsigned k = __float_as_uint(p[2]); int c = __float_as_int(p[3]);
      if (k > bk || (k == bk && c < bc)) { bk = k; bc = c; }
    }
    float phi = phiArr[row];
    float posc = fminf(1.f, fmaxf(-1.f, phi + BETA_F));
    float maxv = fmaxf(unfkey(bk), phi);
    bool mask = (maxv - phi) > 0.f;
    float ps = -phi + logf(se) / ALPHA_F;
    if (mask) { losss += ps; maskc += 1.f; }
    if (bc == targets[row]) precc += 1.f;
    poss += posc; costot += cst;
  }
  float simpos = 0.f, simtot = 0.f, simcnt = 0.f;
  for (int i = tid; i < SIM_TILES * SIM_TILES; i += 256) {
    simpos += simpart[i * 3 + 0];
    simtot += simpart[i * 3 + 1];
    simcnt += simpart[i * 3 + 2];
  }
  __shared__ float red[8][256];
  red[0][tid] = losss; red[1][tid] = maskc; red[2][tid] = precc; red[3][tid] = poss;
  red[4][tid] = costot; red[5][tid] = simpos; red[6][tid] = simtot; red[7][tid] = simcnt;
  __syncthreads();
  for (int s = 128; s > 0; s >>= 1) {
    if (tid < s) {
      #pragma unroll
      for (int q = 0; q < 8; ++q) red[q][tid] += red[q][tid + s];
    }
    __syncthreads();
  }
  if (tid == 0) {
    float L = red[0][0], MC = red[1][0], PR = red[2][0], PS = red[3][0];
    float CT = red[4][0], SP = red[5][0], ST = red[6][0], SC = red[7][0];
    out[0] = L / fmaxf(MC, 1.f);
    out[1] = PR / (float)Nn;
    out[2] = PS / (float)Nn;
    out[3] = (CT - PS) / ((float)Nn * (float)(Cc - 1));
    out[4] = SP / SC;
    out[5] = (ST - SP) / ((float)Nn * (float)Nn - SC);
  }
}

// ---------------------------------------------------------------- launch
extern "C" void kernel_launch(void* const* d_in, const int* in_sizes, int n_in,
                              void* d_out, int out_size, void* d_ws, size_t ws_size,
                              hipStream_t stream) {
  (void)in_sizes; (void)n_in; (void)out_size; (void)ws_size;
  const float* inputs  = (const float*)d_in[0];
  const int*   targets = (const int*)d_in[1];
  const float* kmat    = (const float*)d_in[2];
  float* out = (float*)d_out;

  char* ws = (char*)d_ws;
  const size_t szA = (size_t)Nn * Dd * sizeof(unsigned short);  // 16 MiB
  const size_t szB = (size_t)Cc * Dd * sizeof(unsigned short);  // ~39 MiB
  unsigned short* Ahi = (unsigned short*)(ws);
  unsigned short* Alo = (unsigned short*)(ws + szA);
  unsigned short* Bhi = (unsigned short*)(ws + 2 * szA);
  unsigned short* Blo = (unsigned short*)(ws + 2 * szA + szB);
  const size_t szR = (size_t)Nn * NCHUNK * 4 * sizeof(float);
  float* rowstats = (float*)(ws + 2 * szA + 2 * szB);
  float* phiArr   = (float*)(ws + 2 * szA + 2 * szB + szR);
  float* simpart  = (float*)(ws + 2 * szA + 2 * szB + szR + (size_t)Nn * sizeof(float));

  split_bf16_kernel<<<1024, 256, 0, stream>>>(inputs, Ahi, Alo, Nn * Dd);
  split_bf16_kernel<<<1024, 256, 0, stream>>>(kmat, Bhi, Blo, Cc * Dd);
  phi_kernel<<<Nn, 32, 0, stream>>>(inputs, kmat, targets, phiArr);
  gemm1_kernel<<<dim3(Nn / ROWS_BLK, NCHUNK), 128, 0, stream>>>(
      Ahi, Alo, Bhi, Blo, targets, rowstats);
  sim_kernel<<<dim3(SIM_TILES, SIM_TILES), 128, 0, stream>>>(Ahi, Alo, targets, simpart);
  finalize_kernel<<<1, 256, 0, stream>>>(rowstats, phiArr, targets, simpart, out);
}
